// VisionLlamaPyramidBlock_39814346834445
// MI455X (gfx1250) — matrix operand order repaired and checked by reading, not compiled, not run on hardware
//
#include <hip/hip_runtime.h>
#include <hip/hip_bf16.h>
#include <math.h>

typedef __attribute__((ext_vector_type(16))) _Float16 v16h;
typedef __attribute__((ext_vector_type(8)))  float    v8f;

#define SEQ   9216
#define DIM   256
#define HIDN  1024
#define BATCH 8
#define WSZ   512
#define NWIN  18
#define KEXT  9728           // 512-key look-around pad + 9216
#define SELF_VAL (-50000.0f)

static __device__ __forceinline__ v8f wmma_f16(v16h a, v16h b, v8f c) {
  // v_wmma_f32_16x16x32_f16: D = A(16x32,f16) * B(32x16,f16) + C(16x16,f32)
  return __builtin_amdgcn_wmma_f32_16x16x32_f16(false, a, false, b, (short)0, c, false, false);
}

// A fragment (16x32 MxK) from a row-major f16 matrix.
// ISA layout: lanes g=0: V0..3 -> K 0..7, V4..7 -> K 16..23; lanes g=1: +8.
static __device__ __forceinline__ v16h load_a_frag(const _Float16* base, int row0,
                                                   int stride, int kbase, int lane) {
  int M = lane & 15, g = lane >> 4;
  const _Float16* rp = base + (size_t)(row0 + M) * stride + kbase + 8 * g;
  v16h a;
#pragma unroll
  for (int v = 0; v < 8; ++v) {
    int k = ((v < 4) ? 0 : 16) + 2 * (v & 3);
    a[2 * v]     = rp[k];
    a[2 * v + 1] = rp[k + 1];
  }
  return a;
}

// Same, gathering from an f32 source (LDS probability tiles) with cvt to f16.
static __device__ __forceinline__ v16h load_a_frag_f32(const float* base, int row0,
                                                       int stride, int kbase, int lane) {
  int M = lane & 15, g = lane >> 4;
  const float* rp = base + (size_t)(row0 + M) * stride + kbase + 8 * g;
  v16h a;
#pragma unroll
  for (int v = 0; v < 8; ++v) {
    int k = ((v < 4) ? 0 : 16) + 2 * (v & 3);
    a[2 * v]     = (_Float16)rp[k];
    a[2 * v + 1] = (_Float16)rp[k + 1];
  }
  return a;
}

// B fragment (32x16 KxN): B[k][n] = mat[(col0+n)*stride + kbase + k].
// lanes g: elems 0..7 -> K 8g..8g+7, elems 8..15 -> K 16+8g..23+8g -> two aligned 16B loads.
static __device__ __forceinline__ v16h load_b_frag(const _Float16* base, int col0,
                                                   int stride, int kbase, int lane) {
  int n = lane & 15, g = lane >> 4;
  typedef __attribute__((ext_vector_type(8))) _Float16 v8h_;
  const _Float16* p = base + (size_t)(col0 + n) * stride + kbase + 8 * g;
  v8h_ lo = *(const v8h_*)p;
  v8h_ hi = *(const v8h_*)(p + 16);
  return __builtin_shufflevector(lo, hi, 0, 1, 2, 3, 4, 5, 6, 7, 8, 9, 10, 11, 12, 13, 14, 15);
}

// ---------------------------------------------------------------------------
// K0: weight transpose+cvt to f16, zero the look-around pad regions.
// ---------------------------------------------------------------------------
__global__ void __launch_bounds__(256) k_prep(const float* __restrict__ w1,
                                              const float* __restrict__ w2,
                                              const float* __restrict__ w3,
                                              _Float16* __restrict__ w1hT,
                                              _Float16* __restrict__ w2hT,
                                              _Float16* __restrict__ w3hT,
                                              _Float16* __restrict__ khp,
                                              _Float16* __restrict__ vTp) {
  int idx = blockIdx.x * 256 + threadIdx.x;            // 0 .. 262143
  {
    int hcol = idx >> 8, d = idx & 255;                // w1/w2: [HID][DIM] = W^T
    w1hT[(size_t)hcol * DIM + d] = (_Float16)w1[(size_t)d * HIDN + hcol];
    w2hT[(size_t)hcol * DIM + d] = (_Float16)w2[(size_t)d * HIDN + hcol];
  }
  {
    int dout = idx >> 10, hid = idx & 1023;            // w3: [DIM][HID] = W3^T
    w3hT[(size_t)dout * HIDN + hid] = (_Float16)w3[(size_t)hid * DIM + dout];
  }
#pragma unroll
  for (int r = 0; r < 4; ++r) {
    int e = idx * 4 + r;                               // 0 .. 1048575
    int bb = e >> 17;                                  // / (512*256)
    int rem = e & 131071;
    khp[(size_t)bb * KEXT * DIM + rem] = (_Float16)0.0f;        // first 512 key rows
    int dd = rem >> 9, cc = rem & 511;
    vTp[((size_t)bb * DIM + dd) * KEXT + cc] = (_Float16)0.0f;  // first 512 key cols
  }
}

// ---------------------------------------------------------------------------
// K1: proj -> LN -> xpos rotary q/k (+1/sqrt(d) folded into q) -> k L2-norm.
// One 256-thread block per (b, t) row.
// ---------------------------------------------------------------------------
__global__ void __launch_bounds__(256) k_rope(const float* __restrict__ x,
                                              const float* __restrict__ w_proj,
                                              const float* __restrict__ b_proj,
                                              const float* __restrict__ ln_g,
                                              const float* __restrict__ ln_b,
                                              _Float16* __restrict__ qh,
                                              _Float16* __restrict__ khp,
                                              _Float16* __restrict__ vTp) {
  int bt = blockIdx.x;
  int b = bt / SEQ, t = bt - b * SEQ;
  int d = threadIdx.x;

  float s = b_proj[d];
#pragma unroll
  for (int c = 0; c < 3; ++c)
    s += x[((size_t)(b * 3 + c)) * SEQ + t] * w_proj[c * DIM + d];

  __shared__ float r1[8], r2[8];
  __shared__ float xl[DIM];
  int lane = threadIdx.x & 31, wv = threadIdx.x >> 5;

  float p1 = s, p2 = s * s;
#pragma unroll
  for (int m = 16; m >= 1; m >>= 1) { p1 += __shfl_xor(p1, m, 32); p2 += __shfl_xor(p2, m, 32); }
  if (lane == 0) { r1[wv] = p1; r2[wv] = p2; }
  __syncthreads();
  float tot = 0.0f, tot2 = 0.0f;
#pragma unroll
  for (int i = 0; i < 8; ++i) { tot += r1[i]; tot2 += r2[i]; }
  float mean = tot * (1.0f / DIM);
  float var  = tot2 * (1.0f / DIM) - mean * mean;
  float rstd = rsqrtf(var + 1e-5f);
  float xln = (s - mean) * rstd * ln_g[d] + ln_b[d];
  xl[d] = xln;
  __syncthreads();

  float rot = (d < 128) ? -xl[d + 128] : xl[d - 128];
  int i2 = d & 127;
  float fr = (float)t * powf(10000.0f, -(2.0f * (float)i2) / 256.0f);
  float cs = cosf(fr), sn = sinf(fr);
  float bse = (2.0f * (float)i2 + 0.4f * 256.0f) / (1.4f * 256.0f);
  float pw  = ((float)t - 4608.0f) / 256.0f;          // (t - n//2) / (ws//2)
  float sc  = powf(bse, pw);
  float qk = xln * cs + rot * sn;
  float qv = qk * sc * 0.0625f;                        // * d^-0.5
  float kv = qk / sc;

  float q1 = kv * kv;
#pragma unroll
  for (int m = 16; m >= 1; m >>= 1) q1 += __shfl_xor(q1, m, 32);
  if (lane == 0) r2[wv] = q1;
  __syncthreads();
  float ssq = 0.0f;
#pragma unroll
  for (int i = 0; i < 8; ++i) ssq += r2[i];
  float kn = kv / fmaxf(sqrtf(ssq), 1e-12f);

  qh[(size_t)bt * DIM + d] = (_Float16)qv;
  khp[((size_t)b * KEXT + 512 + t) * DIM + d] = (_Float16)kn;
  vTp[((size_t)b * DIM + d) * KEXT + 512 + t] = (_Float16)xln;  // v = normed, transposed
}

// ---------------------------------------------------------------------------
// K2: local attention. One workgroup (8 waves) per 16-query tile of a window.
// Phase A: S = Q K^T (WMMA) + masks -> LDS.  Softmax in LDS.  Phase B: P V (WMMA).
// Epilogue recomputes the K=3 projection for the residual and writes skip2.
// ---------------------------------------------------------------------------
__global__ void __launch_bounds__(256) k_attn(const float* __restrict__ x,
                                              const float* __restrict__ w_proj,
                                              const float* __restrict__ b_proj,
                                              const _Float16* __restrict__ qh,
                                              const _Float16* __restrict__ khp,
                                              const _Float16* __restrict__ vTp,
                                              float* __restrict__ skip2) {
  extern __shared__ char smem[];
  float* simLDS = (float*)smem;                        // 16 x 1024
  float* redLDS = simLDS + 16 * 1024;                  // 16 x 16
  float* rowmax = redLDS + 256;                        // 16
  float* rowsum = rowmax + 16;                         // 16

  int bx = blockIdx.x;
  int b = bx / (NWIN * 32);
  int rem = bx - b * (NWIN * 32);
  int win = rem >> 5;
  int mt  = rem & 31;
  int t0 = win * WSZ + mt * 16;                        // query base position
  int qrow0 = b * SEQ + t0;
  int krow0 = b * KEXT + win * WSZ;                    // padded key row for j=0
  int kcol0 = win * WSZ;                               // padded V^T column for j=0
  const _Float16* vbase = vTp + (size_t)b * DIM * KEXT;

  int tid = threadIdx.x, lane = tid & 31, wv = tid >> 5;
  int n = lane & 15, gg = lane >> 4;

  v16h qa[8];
#pragma unroll
  for (int s = 0; s < 8; ++s) qa[s] = load_a_frag(qh, qrow0, DIM, s * 32, lane);

  // ---- Phase A: S tiles ----
#pragma unroll 1
  for (int it = 0; it < 8; ++it) {
    int j0 = (wv * 8 + it) * 16;
    v8f acc = {};
#pragma unroll
    for (int s = 0; s < 8; ++s) {
      v16h bf = load_b_frag(khp, krow0 + j0, DIM, s * 32, lane);
      acc = wmma_f16(qa[s], bf, acc);
    }
    int j = j0 + n;
    int kt = (win == 0 && j < 512) ? -1 : ((win - 1) * WSZ + j);
#pragma unroll
    for (int r = 0; r < 8; ++r) {
      int qt = t0 + r + 8 * gg;
      float v = acc[r];
      if (kt < 0 || qt < kt)      v = -INFINITY;       // padding / future
      else if (qt == kt)          v = SELF_VAL;        // self
      simLDS[(r + 8 * gg) * 1024 + j] = v;
    }
  }
  __syncthreads();

  // ---- softmax (keep unnormalized exp + row sums) ----
  {
    int row = tid >> 4, sub = tid & 15;
    float mx = -INFINITY;
    for (int kk = sub; kk < 1024; kk += 16) mx = fmaxf(mx, simLDS[row * 1024 + kk]);
    redLDS[row * 16 + sub] = mx;
    __syncthreads();
    if (sub == 0) {
      float m = -INFINITY;
#pragma unroll
      for (int i = 0; i < 16; ++i) m = fmaxf(m, redLDS[row * 16 + i]);
      rowmax[row] = m;
    }
    __syncthreads();
    float rm = rowmax[row];
    float se = 0.0f;
    for (int kk = sub; kk < 1024; kk += 16) {
      float e = __expf(simLDS[row * 1024 + kk] - rm);
      simLDS[row * 1024 + kk] = e;
      se += e;
    }
    redLDS[row * 16 + sub] = se;
    __syncthreads();
    if (sub == 0) {
      float sm = 0.0f;
#pragma unroll
      for (int i = 0; i < 16; ++i) sm += redLDS[row * 16 + i];
      rowsum[row] = sm;
    }
    __syncthreads();
  }

  // ---- Phase B: out = P @ V (each wave: 2 d-tiles) ----
  v8f oacc0 = {}, oacc1 = {};
  int d0 = wv * 32;
#pragma unroll 1
  for (int s = 0; s < 32; ++s) {
    v16h pa = load_a_frag_f32(simLDS, 0, 1024, s * 32, lane);
    v16h bf0 = load_b_frag(vbase, d0,      KEXT, kcol0 + s * 32, lane);
    oacc0 = wmma_f16(pa, bf0, oacc0);
    v16h bf1 = load_b_frag(vbase, d0 + 16, KEXT, kcol0 + s * 32, lane);
    oacc1 = wmma_f16(pa, bf1, oacc1);
  }

#pragma unroll
  for (int u = 0; u < 2; ++u) {
    v8f oa = u ? oacc1 : oacc0;
    int dcol = d0 + u * 16 + n;
#pragma unroll
    for (int r = 0; r < 8; ++r) {
      int rr = r + 8 * gg;
      int qt = t0 + rr;
      float v = oa[r] / rowsum[rr];
      float sk = b_proj[dcol];                          // recompute skip1 (K=3 proj)
#pragma unroll
      for (int c = 0; c < 3; ++c)
        sk += x[((size_t)(b * 3 + c)) * SEQ + qt] * w_proj[c * DIM + dcol];
      skip2[((size_t)b * SEQ + qt) * DIM + dcol] = v + sk;
    }
  }
}

// ---------------------------------------------------------------------------
// K3: fully fused gated FFN + final LN. One workgroup (8 waves) per 64 rows.
// LN(skip2)->X f16 (LDS) -> X@W1,X@W2 (WMMA) -> silu gate -> H f16 (LDS)
// -> H@W3 (WMMA) + b3 + skip2 -> LN -> out.
// ---------------------------------------------------------------------------
__global__ void __launch_bounds__(256) k_ffn(const float* __restrict__ skip2,
                                             const float* __restrict__ ln_g,
                                             const float* __restrict__ ln_b,
                                             const _Float16* __restrict__ w1hT,
                                             const float* __restrict__ b1,
                                             const _Float16* __restrict__ w2hT,
                                             const float* __restrict__ b2,
                                             const _Float16* __restrict__ w3hT,
                                             const float* __restrict__ b3,
                                             float* __restrict__ out) {
  extern __shared__ char smem[];
  _Float16* XLDS = (_Float16*)smem;                    // 64 x 256  (32 KB)
  _Float16* HLDS = XLDS + 64 * 256;                    // 64 x 1024 (128 KB; also f32 staging)
  float* redm = (float*)(HLDS + 64 * 1024);            // 64 x 4
  float* redv = redm + 256;                            // 64 x 4
  float* rowstat = redv + 256;                         // 64 x 2

  size_t row0 = (size_t)blockIdx.x * 64;
  int tid = threadIdx.x, lane = tid & 31, wv = tid >> 5;
  int mt = wv >> 1;

  // ---- stage 1: LN(skip2) -> X f16 ----
  {
    int row = tid >> 2, sub = tid & 3;
    const float* sp = skip2 + (row0 + row) * DIM;
    float* staging = (float*)HLDS;
    float m = 0.0f, ss = 0.0f;
    for (int k = 0; k < 64; ++k) {
      int col = sub * 64 + k;
      float v = sp[col];
      staging[row * DIM + col] = v;
      m += v; ss += v * v;
    }
    redm[row * 4 + sub] = m; redv[row * 4 + sub] = ss;
    __syncthreads();
    if (sub == 0) {
      float tm = 0.0f, ts = 0.0f;
#pragma unroll
      for (int i = 0; i < 4; ++i) { tm += redm[row * 4 + i]; ts += redv[row * 4 + i]; }
      float mean = tm * (1.0f / DIM);
      float var  = ts * (1.0f / DIM) - mean * mean;
      rowstat[row * 2]     = mean;
      rowstat[row * 2 + 1] = rsqrtf(var + 1e-5f);
    }
    __syncthreads();
    float mean = rowstat[row * 2], rstd = rowstat[row * 2 + 1];
    for (int k = 0; k < 64; ++k) {
      int col = sub * 64 + k;
      float v = staging[row * DIM + col];
      XLDS[row * DIM + col] = (_Float16)((v - mean) * rstd * ln_g[col] + ln_b[col]);
    }
    __syncthreads();
  }

  // ---- stage 2: H = silu(X@W1+b1) * (X@W2+b2) ----
  {
    v16h xa[8];
#pragma unroll
    for (int s = 0; s < 8; ++s) xa[s] = load_a_frag(XLDS, mt * 16, DIM, s * 32, lane);
    int nb = (wv & 1) * 32;
    int n = lane & 15, gg = lane >> 4;
#pragma unroll 1
    for (int i = 0; i < 32; ++i) {
      int hc0 = (nb + i) * 16;
      v8f aA = {}, aB = {};
#pragma unroll
      for (int s = 0; s < 8; ++s) {
        v16h bf1 = load_b_frag(w1hT, hc0, DIM, s * 32, lane);
        aA = wmma_f16(xa[s], bf1, aA);
        v16h bf2 = load_b_frag(w2hT, hc0, DIM, s * 32, lane);
        aB = wmma_f16(xa[s], bf2, aB);
      }
      int hcol = hc0 + n;
      float bb1 = b1[hcol], bb2 = b2[hcol];
#pragma unroll
      for (int r = 0; r < 8; ++r) {
        float a  = aA[r] + bb1;
        float g2 = aB[r] + bb2;
        float sw = a / (1.0f + __expf(-a)) * g2;
        HLDS[(mt * 16 + r + 8 * gg) * HIDN + hcol] = (_Float16)sw;
      }
    }
    __syncthreads();
  }

  // ---- stage 3: Y = H@W3 + b3 + skip2 -> LDS f32 ----
  {
    v8f acc[8] = {};
    int nb = (wv & 1) * 8;
#pragma unroll 1
    for (int s = 0; s < 32; ++s) {
      v16h ha = load_a_frag(HLDS, mt * 16, HIDN, s * 32, lane);
#pragma unroll
      for (int i = 0; i < 8; ++i) {
        v16h bf = load_b_frag(w3hT, (nb + i) * 16, HIDN, s * 32, lane);
        acc[i] = wmma_f16(ha, bf, acc[i]);
      }
    }
    __syncthreads();                                   // done reading HLDS; reuse as OLDS
    float* OLDS = (float*)HLDS;
    int n = lane & 15, gg = lane >> 4;
#pragma unroll
    for (int i = 0; i < 8; ++i) {
      int dcol = (nb + i) * 16 + n;
      float bb = b3[dcol];
#pragma unroll
      for (int r = 0; r < 8; ++r) {
        int lrow = mt * 16 + r + 8 * gg;
        OLDS[lrow * DIM + dcol] = acc[i][r] + bb + skip2[(row0 + lrow) * DIM + dcol];
      }
    }
    __syncthreads();
  }

  // ---- stage 4: final LN -> out ----
  {
    int row = tid >> 2, sub = tid & 3;
    const float* OLDS = (const float*)HLDS;
    float m = 0.0f, ss = 0.0f;
    for (int k = 0; k < 64; ++k) {
      int col = sub * 64 + k;
      float v = OLDS[row * DIM + col];
      m += v; ss += v * v;
    }
    redm[row * 4 + sub] = m; redv[row * 4 + sub] = ss;
    __syncthreads();
    if (sub == 0) {
      float tm = 0.0f, ts = 0.0f;
#pragma unroll
      for (int i = 0; i < 4; ++i) { tm += redm[row * 4 + i]; ts += redv[row * 4 + i]; }
      float mean = tm * (1.0f / DIM);
      float var  = ts * (1.0f / DIM) - mean * mean;
      rowstat[row * 2]     = mean;
      rowstat[row * 2 + 1] = rsqrtf(var + 1e-5f);
    }
    __syncthreads();
    float mean = rowstat[row * 2], rstd = rowstat[row * 2 + 1];
    for (int k = 0; k < 64; ++k) {
      int col = sub * 64 + k;
      float v = OLDS[row * DIM + col];
      out[(row0 + row) * DIM + col] = (v - mean) * rstd * ln_g[col] + ln_b[col];
    }
  }
}

// ---------------------------------------------------------------------------
// Workspace layout (bytes, 256-aligned), total ~195 MB:
//   qh   f16 [8][9216][256]          @ 0          (37,748,736)
//   khp  f16 [8][9728][256]          @ 37748736   (39,845,888)
//   vTp  f16 [8][256][9728]          @ 77594624   (39,845,888)
//   skip2 f32 [8][9216][256]         @ 117440512  (75,497,472)
//   w1hT f16 [1024][256]             @ 192937984  (524,288)
//   w2hT f16 [1024][256]             @ 193462272  (524,288)
//   w3hT f16 [256][1024]             @ 193986560  (524,288)
// ---------------------------------------------------------------------------
extern "C" void kernel_launch(void* const* d_in, const int* in_sizes, int n_in,
                              void* d_out, int out_size, void* d_ws, size_t ws_size,
                              hipStream_t stream) {
  const float* x      = (const float*)d_in[0];
  const float* w_proj = (const float*)d_in[1];
  const float* b_proj = (const float*)d_in[2];
  const float* ln_g   = (const float*)d_in[3];
  const float* ln_b   = (const float*)d_in[4];
  const float* w1     = (const float*)d_in[5];
  const float* b1     = (const float*)d_in[6];
  const float* w2     = (const float*)d_in[7];
  const float* b2     = (const float*)d_in[8];
  const float* w3     = (const float*)d_in[9];
  const float* b3     = (const float*)d_in[10];
  float* out = (float*)d_out;
  char* ws = (char*)d_ws;

  _Float16* qh    = (_Float16*)(ws);
  _Float16* khp   = (_Float16*)(ws + 37748736u);
  _Float16* vTp   = (_Float16*)(ws + 77594624u);
  float*    skip2 = (float*)   (ws + 117440512u);
  _Float16* w1hT  = (_Float16*)(ws + 192937984u);
  _Float16* w2hT  = (_Float16*)(ws + 193462272u);
  _Float16* w3hT  = (_Float16*)(ws + 193986560u);

  k_prep<<<1024, 256, 0, stream>>>(w1, w2, w3, w1hT, w2hT, w3hT, khp, vTp);
  k_rope<<<BATCH * SEQ, 256, 0, stream>>>(x, w_proj, b_proj, ln_g, ln_b, qh, khp, vTp);

  size_t lds_attn = (size_t)(16 * 1024 + 256 + 32) * sizeof(float);      // ~66.7 KB
  k_attn<<<BATCH * NWIN * 32, 256, lds_attn, stream>>>(x, w_proj, b_proj,
                                                       qh, khp, vTp, skip2);

  size_t lds_ffn = (size_t)64 * 256 * 2 + (size_t)64 * 1024 * 2
                 + (size_t)(256 + 256 + 128) * sizeof(float);            // ~162.5 KB
  k_ffn<<<(BATCH * SEQ) / 64, 256, lds_ffn, stream>>>(skip2, ln_g, ln_b,
                                                      w1hT, b1, w2hT, b2,
                                                      w3hT, b3, out);
  (void)in_sizes; (void)n_in; (void)out_size; (void)ws_size;
}
